// ExperimentalSelfAttention_21148418966066
// MI455X (gfx1250) — matrix operand order repaired and checked by reading, not compiled, not run on hardware
//
#include <hip/hip_runtime.h>
#include <hip/hip_bf16.h>

// Problem constants (fixed by the reference).
#define DHID   1024
#define SEQ    2048
#define BATCH  2
#define MROWS  (BATCH*SEQ)   // 4096
#define NHEADS 16
#define NKV    4
#define HDIM   64

typedef __attribute__((ext_vector_type(16))) __bf16       v16bf;
typedef __attribute__((ext_vector_type(8)))  float        v8f;
typedef __attribute__((ext_vector_type(8)))  unsigned int u32x8;

// fp32 -> bf16, round-to-nearest-even.
__device__ __forceinline__ unsigned short f2bf(float f) {
  unsigned u = __float_as_uint(f);
  return (unsigned short)((u + 0x7FFFu + ((u >> 16) & 1u)) >> 16);
}
__device__ __forceinline__ unsigned pack2(unsigned short lo, unsigned short hi) {
  return (unsigned)lo | ((unsigned)hi << 16);
}

__device__ __forceinline__ u32x8 mk8(uint4 a, uint4 b) {
  u32x8 r;
  r[0] = a.x; r[1] = a.y; r[2] = a.z; r[3] = a.w;
  r[4] = b.x; r[5] = b.y; r[6] = b.z; r[7] = b.w;
  return r;
}

// D = A(16x32 bf16) * B(32x16 bf16) + C(16x16 f32), fp32 accumulate.
__device__ __forceinline__ v8f wmma_bf16(u32x8 a, u32x8 b, v8f c) {
  return __builtin_amdgcn_wmma_f32_16x16x32_bf16(
      false, __builtin_bit_cast(v16bf, a),
      false, __builtin_bit_cast(v16bf, b),
      (short)0, c, false, false);
}

// Async LDS DMA: 16 bytes global -> LDS, tracked by ASYNCcnt.
// VDST VGPR carries the LDS byte offset (low 32 bits of a generic LDS pointer,
// per the LDS aperture mapping addr[31:0]).
__device__ __forceinline__ void async_copy_b128(unsigned lds_off, const void* g) {
  asm volatile("global_load_async_to_lds_b128 %0, %1, off"
               :: "v"(lds_off), "v"(g) : "memory");
}
__device__ __forceinline__ void wait_async0() {
  asm volatile("s_wait_asynccnt 0x0" ::: "memory");
}
__device__ __forceinline__ unsigned lds_off32(const void* p) {
  return (unsigned)(unsigned long long)p;  // low 32 bits = LDS byte address
}

// ---------------------------------------------------------------------------
// Kernel 1: QKV projection GEMM, fp32 in -> bf16 out [b][head][s][64].
// Fused bias + optional RoPE + optional 1/sqrt(d) scale.
// Block = 128 threads = 4 waves. Block tile 128(M) x 64(N); wave tile 32x64
// (2 A-fragments x 4 B-fragments = 8 WMMAs per 32-deep K-step).
// ---------------------------------------------------------------------------
__global__ __launch_bounds__(128)
void qkv_gemm(const float* __restrict__ X, const float* __restrict__ W,
              const float* __restrict__ bias, unsigned short* __restrict__ out,
              int N, int heads, int do_rope, float scale)
{
  __shared__ __align__(16) unsigned short Xl[128][32]; // [m][k] bf16
  __shared__ __align__(16) unsigned short Wt[64][32];  // [n][k] bf16 (transposed)
  const int t    = threadIdx.x;
  const int lane = t & 31;
  const int w    = t >> 5;
  const int n    = lane & 15;
  const int half = lane >> 4;
  const int n0   = blockIdx.x * 64;
  const int r0   = blockIdx.y * 128;
  const int head = blockIdx.x;          // 64-wide N tile == one head
  const int cn   = t & 63, kh = t >> 6; // W staging split

  v8f acc[2][4];
  for (int c = 0; c < 4; ++c) {
    float bc = bias[n0 + c*16 + n];
    for (int a = 0; a < 2; ++a)
      for (int i = 0; i < 8; ++i) acc[a][c][i] = bc;
  }

  for (int kk = 0; kk < DHID; kk += 32) {
    // X tile: coalesced float4 loads, packed bf16 uint2 stores.
    for (int u = 0; u < 8; ++u) {
      int f  = u*512 + t*4;
      int xr = f >> 5, xc = f & 31;
      const float4 v4 = *(const float4*)&X[(size_t)(r0 + xr)*DHID + kk + xc];
      uint2 p;
      p.x = pack2(f2bf(v4.x), f2bf(v4.y));
      p.y = pack2(f2bf(v4.z), f2bf(v4.w));
      *(uint2*)&Xl[xr][xc] = p;
    }
    // W tile, transposed into [n][k]: lane-coalesced b32 loads along n.
    for (int u = 0; u < 4; ++u) {
      int kr = kh*16 + u*4;
      float w0 = W[(size_t)(kk + kr + 0)*N + n0 + cn];
      float w1 = W[(size_t)(kk + kr + 1)*N + n0 + cn];
      float w2 = W[(size_t)(kk + kr + 2)*N + n0 + cn];
      float w3 = W[(size_t)(kk + kr + 3)*N + n0 + cn];
      uint2 p;
      p.x = pack2(f2bf(w0), f2bf(w1));
      p.y = pack2(f2bf(w2), f2bf(w3));
      *(uint2*)&Wt[cn][kr] = p;
    }
    __syncthreads();

    u32x8 af[2];
    for (int a = 0; a < 2; ++a) {
      const uint4* xrow = (const uint4*)&Xl[w*32 + a*16 + n][0];
      af[a] = mk8(xrow[half], xrow[2 + half]);
    }
    for (int c = 0; c < 4; ++c) {
      const uint4* wrow = (const uint4*)&Wt[c*16 + n][0];
      u32x8 bfrag = mk8(wrow[half], wrow[2 + half]);
      acc[0][c] = wmma_bf16(af[0], bfrag, acc[0][c]);
      acc[1][c] = wmma_bf16(af[1], bfrag, acc[1][c]);
    }
    __syncthreads();
  }

  // Epilogue: scale + RoPE (partner (d, d+32) = fragments c and c+2, same lane).
  float invf[2];
  invf[0] = __powf(10000.0f, -(float)(n)      * (1.0f/32.0f));
  invf[1] = __powf(10000.0f, -(float)(16 + n) * (1.0f/32.0f));
  for (int a = 0; a < 2; ++a) {
    for (int i = 0; i < 8; ++i) {
      int rr = r0 + w*32 + a*16 + i + 8*half;
      int b  = rr >> 11;              // / SEQ
      int s  = rr & (SEQ - 1);
      float o[4];
      for (int c = 0; c < 4; ++c) o[c] = acc[a][c][i] * scale;
      if (do_rope) {
        for (int c = 0; c < 2; ++c) {
          float sn, cs;
          __sincosf((float)s * invf[c], &sn, &cs);
          float x1 = o[c], x2 = o[c+2];
          o[c]   = x1*cs - x2*sn;
          o[c+2] = x1*sn + x2*cs;
        }
      }
      unsigned short* orow = out + ((size_t)(b*heads + head)*SEQ + s)*HDIM;
      for (int c = 0; c < 4; ++c) orow[c*16 + n] = f2bf(o[c]);
    }
  }
}

// ---------------------------------------------------------------------------
// Kernel 2: flash attention. One block per (b, head, 64-query tile); 4 waves x 16 queries.
// K tile staged with GLOBAL_LOAD_ASYNC_TO_LDS_B128 (ASYNCcnt); V^T via
// vectorized register transpose. Online softmax fp32; bf16 WMMA throughout.
// ---------------------------------------------------------------------------
__global__ __launch_bounds__(128)
void flash_attn(const unsigned short* __restrict__ Q,
                const unsigned short* __restrict__ Kb,
                const unsigned short* __restrict__ Vb,
                const float* __restrict__ mask,
                unsigned short* __restrict__ Ob)
{
  __shared__ __align__(16) unsigned short Kl[32][64];     // [key][dim]
  __shared__ __align__(16) unsigned short Vt[64][32];     // [dim][key]
  __shared__ __align__(16) unsigned short Pl[4][16][32];  // per-wave P patch
  const int t    = threadIdx.x, lane = t & 31, w = t >> 5;
  const int n    = lane & 15, half = lane >> 4;
  const int b    = blockIdx.x >> 4, h = blockIdx.x & 15, kv = h >> 2;
  const int q0   = blockIdx.y * 64 + w * 16;

  // Q fragments (16 queries x 64 dims), resident in VGPRs for the whole sweep.
  const unsigned short* qrow = Q + ((size_t)(b*NHEADS + h)*SEQ + q0 + n)*HDIM;
  u32x8 qa[2];
  for (int ks = 0; ks < 2; ++ks) {
    const uint4* p = (const uint4*)(qrow + ks*32);
    qa[ks] = mk8(p[half], p[2 + half]);
  }

  v8f o[4];
  for (int c = 0; c < 4; ++c) for (int i = 0; i < 8; ++i) o[c][i] = 0.0f;
  float mi[8], li[8];
  for (int i = 0; i < 8; ++i) { mi[i] = -1e30f; li[i] = 0.0f; }

  const unsigned short* Kbase = Kb + (size_t)(b*NKV + kv)*SEQ*HDIM;
  const unsigned short* Vbase = Vb + (size_t)(b*NKV + kv)*SEQ*HDIM;
  const float* mrow = mask + (size_t)b*SEQ;

  // Per-thread staging geometry.
  const int kky = t >> 2, kkd = (t & 3) * 16;   // K async: 32B per thread
  for (int kt = 0; kt < SEQ; kt += 32) {
    // K tile: async DMA straight into LDS (no VGPR round-trip).
    {
      const unsigned short* g = Kbase + (size_t)(kt + kky)*HDIM + kkd;
      unsigned lo = lds_off32(&Kl[kky][kkd]);
      async_copy_b128(lo,      g);
      async_copy_b128(lo + 16, g + 8);
    }
    // V tile: vector global reads, transposed scatter into LDS.
    for (int u = 0; u < 4; ++u) {
      int f  = u*512 + t*4;
      int ky = f >> 6, dd = f & 63;
      uint2 v2 = *(const uint2*)&Vbase[(size_t)(kt + ky)*HDIM + dd];
      Vt[dd + 0][ky] = (unsigned short)(v2.x & 0xFFFF);
      Vt[dd + 1][ky] = (unsigned short)(v2.x >> 16);
      Vt[dd + 2][ky] = (unsigned short)(v2.y & 0xFFFF);
      Vt[dd + 3][ky] = (unsigned short)(v2.y >> 16);
    }
    if (kt + 32 < SEQ) {  // pull next tile toward L2 (global_prefetch_b8)
      __builtin_prefetch(Kbase + (size_t)(kt + 32 + (t >> 2))*HDIM, 0, 1);
      __builtin_prefetch(Vbase + (size_t)(kt + 32 + (t >> 2))*HDIM, 0, 1);
    }
    wait_async0();
    __syncthreads();

    // Scores: S(16x32) = Q(16x64) x K_tile^T as two 16x16 C-fragments.
    v8f sc[2];
    for (int ch = 0; ch < 2; ++ch) {
      v8f z; for (int i = 0; i < 8; ++i) z[i] = 0.0f;
      const uint4* kp = (const uint4*)&Kl[ch*16 + n][0];
      for (int ks = 0; ks < 2; ++ks) {
        u32x8 kbf = mk8(kp[ks*4 + half], kp[ks*4 + 2 + half]);
        z = wmma_bf16(qa[ks], kbf, z);
      }
      float mb = (1.0f - mrow[kt + ch*16 + n]) * -1e9f;  // mask bias
      for (int i = 0; i < 8; ++i) z[i] += mb;
      sc[ch] = z;
    }

    // Online softmax; row m = i + 8*half striped over 16 lanes (cols).
    for (int i = 0; i < 8; ++i) {
      float mx = fmaxf(sc[0][i], sc[1][i]);
      mx = fmaxf(mx, __shfl_xor(mx, 1, 32));
      mx = fmaxf(mx, __shfl_xor(mx, 2, 32));
      mx = fmaxf(mx, __shfl_xor(mx, 4, 32));
      mx = fmaxf(mx, __shfl_xor(mx, 8, 32));
      float mnew  = fmaxf(mi[i], mx);
      float alpha = __expf(mi[i] - mnew);
      float p0 = __expf(sc[0][i] - mnew);
      float p1 = __expf(sc[1][i] - mnew);
      float ls = p0 + p1;
      ls += __shfl_xor(ls, 1, 32);
      ls += __shfl_xor(ls, 2, 32);
      ls += __shfl_xor(ls, 4, 32);
      ls += __shfl_xor(ls, 8, 32);
      li[i] = li[i]*alpha + ls;
      mi[i] = mnew;
      sc[0][i] = p0; sc[1][i] = p1;
      for (int c = 0; c < 4; ++c) o[c][i] *= alpha;
    }

    // C-layout -> A-layout for P via per-wave LDS patch (LDS in-order per wave).
    for (int i = 0; i < 8; ++i) {
      int rm = i + 8*half;
      Pl[w][rm][n]      = f2bf(sc[0][i]);
      Pl[w][rm][16 + n] = f2bf(sc[1][i]);
    }
    __builtin_amdgcn_wave_barrier();
    const uint4* pr = (const uint4*)&Pl[w][n][0];
    u32x8 pf = mk8(pr[half], pr[2 + half]);

    // O(16x64) += P(16x32) x V_tile(32x64).
    for (int c = 0; c < 4; ++c) {
      const uint4* vr = (const uint4*)&Vt[c*16 + n][0];
      u32x8 vbf = mk8(vr[half], vr[2 + half]);
      o[c] = wmma_bf16(pf, vbf, o[c]);
    }
    __syncthreads();
  }

  // Normalize, write bf16 [b][s][h*64+d] for the output projection.
  for (int i = 0; i < 8; ++i) {
    float inv = 1.0f / li[i];
    int s = q0 + i + 8*half;
    unsigned short* orow = Ob + ((size_t)b*SEQ + s)*DHID + h*HDIM;
    for (int c = 0; c < 4; ++c) orow[c*16 + n] = f2bf(o[c][i]*inv);
  }
}

// ---------------------------------------------------------------------------
// Kernel 3: output projection. A (bf16) staged via async LDS DMA; W converted
// on load; fp32 out + bias. Block tile 128x64, wave tile 32x64.
// ---------------------------------------------------------------------------
__global__ __launch_bounds__(128)
void out_gemm(const unsigned short* __restrict__ A, const float* __restrict__ W,
              const float* __restrict__ bias, float* __restrict__ out)
{
  __shared__ __align__(16) unsigned short Xl[128][32];
  __shared__ __align__(16) unsigned short Wt[64][32];
  const int t = threadIdx.x, lane = t & 31, w = t >> 5;
  const int n = lane & 15, half = lane >> 4;
  const int n0 = blockIdx.x*64, r0 = blockIdx.y*128;
  const int cn = t & 63, kh = t >> 6;

  v8f acc[2][4];
  for (int c = 0; c < 4; ++c) {
    float bc = bias[n0 + c*16 + n];
    for (int a = 0; a < 2; ++a)
      for (int i = 0; i < 8; ++i) acc[a][c][i] = bc;
  }

  for (int kk = 0; kk < DHID; kk += 32) {
    // A tile: pure bf16 copy -> async DMA to LDS.
    for (int u = 0; u < 2; ++u) {
      int f  = u*2048 + t*16;
      int xr = f >> 5, xc = f & 31;
      const unsigned short* g = &A[(size_t)(r0 + xr)*DHID + kk + xc];
      unsigned lo = lds_off32(&Xl[xr][xc]);
      async_copy_b128(lo,      g);
      async_copy_b128(lo + 16, g + 8);
    }
    // W tile transposed, with fp32 -> bf16 conversion.
    for (int u = 0; u < 4; ++u) {
      int kr = kh*16 + u*4;
      float w0 = W[(size_t)(kk + kr + 0)*DHID + n0 + cn];
      float w1 = W[(size_t)(kk + kr + 1)*DHID + n0 + cn];
      float w2 = W[(size_t)(kk + kr + 2)*DHID + n0 + cn];
      float w3 = W[(size_t)(kk + kr + 3)*DHID + n0 + cn];
      uint2 p;
      p.x = pack2(f2bf(w0), f2bf(w1));
      p.y = pack2(f2bf(w2), f2bf(w3));
      *(uint2*)&Wt[cn][kr] = p;
    }
    wait_async0();
    __syncthreads();

    u32x8 af[2];
    for (int a = 0; a < 2; ++a) {
      const uint4* xrow = (const uint4*)&Xl[w*32 + a*16 + n][0];
      af[a] = mk8(xrow[half], xrow[2 + half]);
    }
    for (int c = 0; c < 4; ++c) {
      const uint4* wrow = (const uint4*)&Wt[c*16 + n][0];
      u32x8 bfrag = mk8(wrow[half], wrow[2 + half]);
      acc[0][c] = wmma_bf16(af[0], bfrag, acc[0][c]);
      acc[1][c] = wmma_bf16(af[1], bfrag, acc[1][c]);
    }
    __syncthreads();
  }

  for (int a = 0; a < 2; ++a) {
    for (int i = 0; i < 8; ++i) {
      int rr = r0 + w*32 + a*16 + i + 8*half;
      float* orow = out + (size_t)rr*DHID + n0;
      for (int c = 0; c < 4; ++c) orow[c*16 + n] = acc[a][c][i];
    }
  }
}

// ---------------------------------------------------------------------------
extern "C" void kernel_launch(void* const* d_in, const int* in_sizes, int n_in,
                              void* d_out, int out_size, void* d_ws, size_t ws_size,
                              hipStream_t stream)
{
  const float* X    = (const float*)d_in[0];
  const float* mask = (const float*)d_in[1];
  const float* Wq   = (const float*)d_in[2];
  const float* bq   = (const float*)d_in[3];
  const float* Wk   = (const float*)d_in[4];
  const float* bk   = (const float*)d_in[5];
  const float* Wv   = (const float*)d_in[6];
  const float* bv   = (const float*)d_in[7];
  const float* Wo   = (const float*)d_in[8];
  const float* bo   = (const float*)d_in[9];
  float* out = (float*)d_out;

  // Workspace: q(8MB) | k(2MB) | v(2MB) | attn_out(8MB), all bf16.
  char* ws = (char*)d_ws;
  unsigned short* qb = (unsigned short*)(ws);
  unsigned short* kb = (unsigned short*)(ws + (size_t)MROWS*DHID*2);
  unsigned short* vb = (unsigned short*)(ws + (size_t)MROWS*DHID*2 + (size_t)MROWS*256*2);
  unsigned short* ab = (unsigned short*)(ws + (size_t)MROWS*DHID*2 + (size_t)MROWS*256*4);

  dim3 blk(128);
  // Q: fold 1/sqrt(64) into the projection; RoPE fused. K: RoPE. V: plain.
  qkv_gemm<<<dim3(16, MROWS/128), blk, 0, stream>>>(X, Wq, bq, qb, 1024, NHEADS, 1, 0.125f);
  qkv_gemm<<<dim3(4,  MROWS/128), blk, 0, stream>>>(X, Wk, bk, kb, 256,  NKV,    1, 1.0f);
  qkv_gemm<<<dim3(4,  MROWS/128), blk, 0, stream>>>(X, Wv, bv, vb, 256,  NKV,    0, 1.0f);
  flash_attn<<<dim3(BATCH*NHEADS, SEQ/64), blk, 0, stream>>>(qb, kb, vb, mask, ab);
  out_gemm<<<dim3(16, MROWS/128), blk, 0, stream>>>(ab, Wo, bo, out);
}
